// TFHindiCausalLMAttention_72078141161498
// MI455X (gfx1250) — matrix operand order repaired and checked by reading, not compiled, not run on hardware
//
#include <hip/hip_runtime.h>
#include <hip/hip_bf16.h>

typedef __attribute__((ext_vector_type(16))) _Float16 v16h;
typedef __attribute__((ext_vector_type(8)))  _Float16 v8h;
typedef __attribute__((ext_vector_type(8)))  float    v8f;

#define SEQ    2048
#define HID    2048
#define NH     32
#define NKV    8
#define HD     64
#define KVW    (NKV * HD)       /* 512 */
#define NEGBIG (-1.0e9f)

// ---------------------------------------------------------------------------
// WMMA fragment loaders (CDNA5 16-bit layouts, CDNA5 ISA §7.12.2)
// A (16x32, MxK): lane m = lane&15; kbase=(lane>>4)*8; half h -> K = (h<8 ? kbase+h : 16+kbase+(h-8))
// B (32x16, KxN) from BT[N,K] row-major: lane n = lane&15; kbase=(lane>>4)*8; half h -> K = (h<8 ? kbase+h : 16+kbase+(h-8))
// C/D (16x16 f32): lane col n = lane&15; vgpr r -> row r + 8*(lane>>4)
// ---------------------------------------------------------------------------
__device__ __forceinline__ v16h load_a_frag(const _Float16* A, int lda, int m0, int k0, int lane) {
  const int m  = m0 + (lane & 15);
  const int kb = (lane >> 4) << 3;          // 0 or 8
  const _Float16* p = A + (size_t)m * lda + k0 + kb;
  v8h lo = *(const v8h*)(p);                // K = k0+kb .. +7
  v8h hi = *(const v8h*)(p + 16);           // K = k0+16+kb .. +7
  v16h r;
#pragma unroll
  for (int i = 0; i < 8; ++i) { r[i] = lo[i]; r[i + 8] = hi[i]; }
  return r;
}

__device__ __forceinline__ v16h load_b_frag(const _Float16* BT, int ldb, int n0, int k0, int lane) {
  const int n  = n0 + (lane & 15);
  const int kb = (lane >> 4) << 3;          // 0 or 8
  const _Float16* p = BT + (size_t)n * ldb + k0 + kb;
  v8h lo = *(const v8h*)(p);
  v8h hi = *(const v8h*)(p + 16);
  v16h r;
#pragma unroll
  for (int i = 0; i < 8; ++i) { r[i] = lo[i]; r[i + 8] = hi[i]; }
  return r;
}

#define WMMA_F32_F16(a, b, c) \
  __builtin_amdgcn_wmma_f32_16x16x32_f16(false, (a), false, (b), (short)0, (c), false, false)

// ---------------------------------------------------------------------------
// Generic f16 GEMM: C[M,N] = A[M,K] * BT[N,K]^T.
// Wave tile 32x32 (4 accumulators) -- low register pressure so the compiler
// 4x-unrolls the K-loop and issues staggered s_wait_loadcnt (overlap loads
// with WMMA).  8 waves/block arranged 2x4 -> block tile 64x128.
// ---------------------------------------------------------------------------
__global__ void __launch_bounds__(256) gemm_f16f32(const _Float16* __restrict__ A,
                                                   const _Float16* __restrict__ BT,
                                                   float* __restrict__ C,
                                                   int M, int N, int K) {
  const int lane = threadIdx.x & 31;
  const int wave = threadIdx.x >> 5;
  const int m0 = blockIdx.y * 64  + (wave >> 2) * 32;
  const int n0 = blockIdx.x * 128 + (wave & 3) * 32;
  v8f acc00 = {}, acc01 = {}, acc10 = {}, acc11 = {};
#pragma unroll 4
  for (int k0 = 0; k0 < K; k0 += 32) {
    // prefetch ~4 K-steps ahead (gfx1250 global_prefetch_b8; wave-uniform guard)
    if (k0 + 128 < K) {
      __builtin_prefetch(A  + (size_t)(m0 + (lane & 15)) * K + k0 + 128, 0, 1);
      __builtin_prefetch(BT + (size_t)(n0 + (lane & 15)) * K + k0 + 128, 0, 1);
    }
    v16h a0 = load_a_frag(A, K, m0,      k0, lane);
    v16h a1 = load_a_frag(A, K, m0 + 16, k0, lane);
    v16h b0 = load_b_frag(BT, K, n0,      k0, lane);
    v16h b1 = load_b_frag(BT, K, n0 + 16, k0, lane);
    acc00 = WMMA_F32_F16(a0, b0, acc00);
    acc01 = WMMA_F32_F16(a0, b1, acc01);
    acc10 = WMMA_F32_F16(a1, b0, acc10);
    acc11 = WMMA_F32_F16(a1, b1, acc11);
  }
  const int roff = (lane >> 4) << 3;
  const int c    = lane & 15;
#pragma unroll
  for (int r = 0; r < 8; ++r) {
    C[(size_t)(m0 + roff + r)      * N + n0 + c]      = acc00[r];
    C[(size_t)(m0 + roff + r)      * N + n0 + 16 + c] = acc01[r];
    C[(size_t)(m0 + 16 + roff + r) * N + n0 + c]      = acc10[r];
    C[(size_t)(m0 + 16 + roff + r) * N + n0 + 16 + c] = acc11[r];
  }
}

// ---------------------------------------------------------------------------
// Elementwise helpers
// ---------------------------------------------------------------------------
__global__ void cvt_f32_f16(const float* __restrict__ s, _Float16* __restrict__ d, int n) {
  int i = blockIdx.x * 256 + threadIdx.x;
  if (i < n) d[i] = (_Float16)s[i];
}

// D[c*R + r] = (f16) S[r*C + c]
__global__ void transpose_cvt(const float* __restrict__ S, _Float16* __restrict__ D,
                              int R, int Ccols) {
  int c = blockIdx.x * 256 + threadIdx.x;
  int r = blockIdx.y;
  if (c < Ccols) D[(size_t)c * R + r] = (_Float16)S[(size_t)r * Ccols + c];
}

// RoPE (per head of 64) + optional 1/sqrt(HD) scale, f32 -> f16
__global__ void rope_cvt(const float* __restrict__ S, _Float16* __restrict__ D,
                         const int* __restrict__ pos, int ld, float scale) {
  int col = blockIdx.x * 256 + threadIdx.x;
  int s = blockIdx.y;
  if (col >= ld) return;
  int d = col & (HD - 1);
  int i = d & 31;
  float p = (float)pos[s];
  float freq = __powf(10000.0f, -(float)i * (1.0f / 32.0f));
  float ang = p * freq;
  float sn, cs;
  __sincosf(ang, &sn, &cs);
  size_t base = (size_t)s * ld;
  float x = S[base + col];
  float other = (d < 32) ? -S[base + col + 32] : S[base + col - 32];
  D[base + col] = (_Float16)((x * cs + other * sn) * scale);
}

// ---------------------------------------------------------------------------
// Flash attention: grid (S/128, NH), 8 waves/block, wave = 16 query rows x full HD.
// Qh [S,2048] f16 (pre-scaled), Kh [S,512] f16 (rope'd), VT [512,S] f16, Oh [S,2048] f16.
// ---------------------------------------------------------------------------
__global__ void __launch_bounds__(256) flash_attn(const _Float16* __restrict__ Qh,
                                                  const _Float16* __restrict__ Kh,
                                                  const _Float16* __restrict__ VT,
                                                  _Float16* __restrict__ Oh) {
  __shared__ __align__(16) _Float16 plds[8][16 * 32];
  const int lane = threadIdx.x & 31;
  const int wave = threadIdx.x >> 5;
  const int head = blockIdx.y;
  const int kvh  = head >> 2;                 // GQA group of 4
  const int q0   = blockIdx.x * 128 + wave * 16;
  const int roff = (lane >> 4) << 3;          // C-layout row offset per half-wave
  const int cin  = lane & 15;

  v16h qf0 = load_a_frag(Qh + head * HD, HID, q0, 0,  lane);
  v16h qf1 = load_a_frag(Qh + head * HD, HID, q0, 32, lane);

  v8f o0 = {}, o1 = {}, o2 = {}, o3 = {};
  float rm[8], rl[8];
#pragma unroll
  for (int r = 0; r < 8; ++r) { rm[r] = -3.0e38f; rl[r] = 0.0f; }

  _Float16* pb = &plds[wave][0];
  const _Float16* Kb = Kh + kvh * HD;
  const _Float16* Vb = VT + (size_t)(kvh * HD) * SEQ;

  for (int j0 = 0; j0 < q0 + 16; j0 += 32) {   // causal: only key blocks <= query tile
    v8f s0 = {}, s1 = {};
    v16h k00 = load_b_frag(Kb, KVW, j0,      0,  lane);
    v16h k01 = load_b_frag(Kb, KVW, j0,      32, lane);
    v16h k10 = load_b_frag(Kb, KVW, j0 + 16, 0,  lane);
    v16h k11 = load_b_frag(Kb, KVW, j0 + 16, 32, lane);
    s0 = WMMA_F32_F16(qf0, k00, s0);
    s0 = WMMA_F32_F16(qf1, k01, s0);
    s1 = WMMA_F32_F16(qf0, k10, s1);
    s1 = WMMA_F32_F16(qf1, k11, s1);

    // causal mask (element (row, col) with col = j0 + t*16 + cin)
#pragma unroll
    for (int r = 0; r < 8; ++r) {
      int row = q0 + roff + r;
      if (j0 + cin      > row) s0[r] = NEGBIG;
      if (j0 + 16 + cin > row) s1[r] = NEGBIG;
    }

    // row max over the 16 lanes holding this row's columns
    float bm[8];
#pragma unroll
    for (int r = 0; r < 8; ++r) bm[r] = fmaxf(s0[r], s1[r]);
#pragma unroll
    for (int off = 1; off < 16; off <<= 1) {
#pragma unroll
      for (int r = 0; r < 8; ++r) bm[r] = fmaxf(bm[r], __shfl_xor(bm[r], off, 32));
    }

    float alpha[8], rs[8];
#pragma unroll
    for (int r = 0; r < 8; ++r) {
      float nm = fmaxf(rm[r], bm[r]);
      alpha[r] = __expf(rm[r] - nm);
      rm[r] = nm;
      float p0 = __expf(s0[r] - nm);
      float p1 = __expf(s1[r] - nm);
      s0[r] = p0; s1[r] = p1;
      rs[r] = p0 + p1;
    }
#pragma unroll
    for (int off = 1; off < 16; off <<= 1) {
#pragma unroll
      for (int r = 0; r < 8; ++r) rs[r] += __shfl_xor(rs[r], off, 32);
    }
#pragma unroll
    for (int r = 0; r < 8; ++r) {
      rl[r] = rl[r] * alpha[r] + rs[r];
      o0[r] *= alpha[r]; o1[r] *= alpha[r]; o2[r] *= alpha[r]; o3[r] *= alpha[r];
      pb[(roff + r) * 32 + cin]      = (_Float16)s0[r];   // P tile 16x32 row-major
      pb[(roff + r) * 32 + 16 + cin] = (_Float16)s1[r];
    }
    __builtin_amdgcn_fence(__ATOMIC_ACQ_REL, "wavefront");  // order LDS store -> load (same wave)

    v16h pf = load_a_frag(pb, 32, 0, 0, lane);              // P as A-fragment
    v16h v0 = load_b_frag(Vb, SEQ, 0,  j0, lane);
    v16h v1 = load_b_frag(Vb, SEQ, 16, j0, lane);
    v16h v2 = load_b_frag(Vb, SEQ, 32, j0, lane);
    v16h v3 = load_b_frag(Vb, SEQ, 48, j0, lane);
    o0 = WMMA_F32_F16(pf, v0, o0);
    o1 = WMMA_F32_F16(pf, v1, o1);
    o2 = WMMA_F32_F16(pf, v2, o2);
    o3 = WMMA_F32_F16(pf, v3, o3);
    __builtin_amdgcn_fence(__ATOMIC_ACQ_REL, "wavefront");  // before next iter rewrites pb
  }

#pragma unroll
  for (int r = 0; r < 8; ++r) {
    float inv = 1.0f / rl[r];
    size_t rowbase = (size_t)(q0 + roff + r) * HID + head * HD;
    Oh[rowbase + cin]      = (_Float16)(o0[r] * inv);
    Oh[rowbase + 16 + cin] = (_Float16)(o1[r] * inv);
    Oh[rowbase + 32 + cin] = (_Float16)(o2[r] * inv);
    Oh[rowbase + 48 + cin] = (_Float16)(o3[r] * inv);
  }
}

// ---------------------------------------------------------------------------
extern "C" void kernel_launch(void* const* d_in, const int* in_sizes, int n_in,
                              void* d_out, int out_size, void* d_ws, size_t ws_size,
                              hipStream_t stream) {
  const float* X   = (const float*)d_in[0];
  // d_in[1] = attention_mask: unused (causal mask recomputed inline)
  const int*   pos = (const int*)d_in[2];
  const float* Wq  = (const float*)d_in[3];
  const float* Wk  = (const float*)d_in[4];
  const float* Wv  = (const float*)d_in[5];
  const float* Wo  = (const float*)d_in[6];
  float* out = (float*)d_out;

  char* ws = (char*)d_ws;
  size_t off = 0;
  _Float16* Xh  = (_Float16*)(ws + off); off += (size_t)SEQ * HID * 2;   // X in f16
  _Float16* WqT = (_Float16*)(ws + off); off += (size_t)HID * HID * 2;   // Wq^T f16
  _Float16* WkT = (_Float16*)(ws + off); off += (size_t)KVW * HID * 2;
  _Float16* WvT = (_Float16*)(ws + off); off += (size_t)KVW * HID * 2;
  _Float16* WoT = (_Float16*)(ws + off); off += (size_t)HID * HID * 2;
  float*    Qf  = (float*)(ws + off);    off += (size_t)SEQ * HID * 4;   // Q proj f32
  float*    Kf  = (float*)(ws + off);    off += (size_t)SEQ * KVW * 4;
  float*    Vf  = (float*)(ws + off);    off += (size_t)SEQ * KVW * 4;
  _Float16* Qh  = (_Float16*)(ws + off); off += (size_t)SEQ * HID * 2;   // rope(Q)/8
  _Float16* Khh = (_Float16*)(ws + off); off += (size_t)SEQ * KVW * 2;   // rope(K)
  _Float16* VT  = (_Float16*)(ws + off); off += (size_t)KVW * SEQ * 2;   // V^T
  _Float16* Oh  = (_Float16*)(ws + off); off += (size_t)SEQ * HID * 2;   // attn out f16

  // 1) precision conversion / weight transposition
  cvt_f32_f16<<<dim3(SEQ * HID / 256), 256, 0, stream>>>(X, Xh, SEQ * HID);
  transpose_cvt<<<dim3(HID / 256, HID), 256, 0, stream>>>(Wq, WqT, HID, HID);
  transpose_cvt<<<dim3(KVW / 256, HID), 256, 0, stream>>>(Wk, WkT, HID, KVW);
  transpose_cvt<<<dim3(KVW / 256, HID), 256, 0, stream>>>(Wv, WvT, HID, KVW);
  transpose_cvt<<<dim3(HID / 256, HID), 256, 0, stream>>>(Wo, WoT, HID, HID);

  // 2) QKV projections (WMMA)
  gemm_f16f32<<<dim3(HID / 128, SEQ / 64), 256, 0, stream>>>(Xh, WqT, Qf, SEQ, HID, HID);
  gemm_f16f32<<<dim3(KVW / 128, SEQ / 64), 256, 0, stream>>>(Xh, WkT, Kf, SEQ, KVW, HID);
  gemm_f16f32<<<dim3(KVW / 128, SEQ / 64), 256, 0, stream>>>(Xh, WvT, Vf, SEQ, KVW, HID);

  // 3) RoPE (+1/sqrt(64) folded into Q) and V transpose, all -> f16
  rope_cvt<<<dim3(HID / 256, SEQ), 256, 0, stream>>>(Qf, Qh, pos, HID, 0.125f);
  rope_cvt<<<dim3(KVW / 256, SEQ), 256, 0, stream>>>(Kf, Khh, pos, KVW, 1.0f);
  transpose_cvt<<<dim3(KVW / 256, SEQ), 256, 0, stream>>>(Vf, VT, SEQ, KVW);

  // 4) causal flash attention (WMMA for QK^T and PV)
  flash_attn<<<dim3(SEQ / 128, NH), 256, 0, stream>>>(Qh, Khh, VT, Oh);

  // 5) output projection (WMMA) -> f32 d_out
  gemm_f16f32<<<dim3(HID / 128, SEQ / 64), 256, 0, stream>>>(Oh, WoT, out, SEQ, HID, HID);
}
